// SparseMoELayer_67448166416810
// MI455X (gfx1250) — matrix operand order repaired and checked by reading, not compiled, not run on hardware
//
#include <hip/hip_runtime.h>

// ---------------------------------------------------------------------------
// SparseMoE (top-2 of 8 experts + shared expert, SwiGLU) for gfx1250 (MI455X).
// bf16 WMMA GEMMs with f32 accumulate; fp32 weights converted to bf16 in
// registers while staging into LDS.  Double-buffered LDS pipeline with
// GLOBAL_LOAD_ASYNC_TO_LDS for the activation tiles (ASYNCcnt-tracked).
// Requires ~140.7 MB of workspace.
// ---------------------------------------------------------------------------

typedef __attribute__((ext_vector_type(16))) __bf16 v16bf;
typedef __attribute__((ext_vector_type(8)))  float  v8f;

union BVec { uint4 q[2]; v16bf v; };

constexpr int kH   = 2048;   // hidden
constexpr int kF   = 2048;   // expert intermediate
constexpr int kE   = 8;      // experts
constexpr int kT   = 4096;   // tokens (B*S)
constexpr int PCAP = 8704;   // padded pair capacity: T*K + E*63 rounded to 64

constexpr int BM = 64, BN = 128, BK = 32;   // workgroup tile; 8 waves of 32x32
constexpr int KSTEPS = kH / BK;             // 64 (kH == kF)

__device__ __forceinline__ unsigned short f2bf(float f) {
    unsigned int u = __float_as_uint(f);
    unsigned int r = u + 0x7fffu + ((u >> 16) & 1u);   // round-to-nearest-even
    return (unsigned short)(r >> 16);
}

__device__ __forceinline__ void store_bf16x16(unsigned short* dst,
                                              float4 a, float4 b, float4 c, float4 d) {
    unsigned int u0 = (unsigned int)f2bf(a.x) | ((unsigned int)f2bf(a.y) << 16);
    unsigned int u1 = (unsigned int)f2bf(a.z) | ((unsigned int)f2bf(a.w) << 16);
    unsigned int u2 = (unsigned int)f2bf(b.x) | ((unsigned int)f2bf(b.y) << 16);
    unsigned int u3 = (unsigned int)f2bf(b.z) | ((unsigned int)f2bf(b.w) << 16);
    unsigned int u4 = (unsigned int)f2bf(c.x) | ((unsigned int)f2bf(c.y) << 16);
    unsigned int u5 = (unsigned int)f2bf(c.z) | ((unsigned int)f2bf(c.w) << 16);
    unsigned int u6 = (unsigned int)f2bf(d.x) | ((unsigned int)f2bf(d.y) << 16);
    unsigned int u7 = (unsigned int)f2bf(d.z) | ((unsigned int)f2bf(d.w) << 16);
    ((uint4*)dst)[0] = make_uint4(u0, u1, u2, u3);
    ((uint4*)dst)[1] = make_uint4(u4, u5, u6, u7);
}

// Async copy 16B/lane from global to LDS (VGLOBAL encoding, ASYNCcnt-tracked).
__device__ __forceinline__ void async_ld_b128(unsigned lds_byte_addr, const void* gaddr) {
    asm volatile("global_load_async_to_lds_b128 %0, %1, off"
                 :: "v"(lds_byte_addr), "v"(gaddr) : "memory");
}
__device__ __forceinline__ void wait_asynccnt0() {
    asm volatile("s_wait_asynccnt 0x0" ::: "memory");
}

// ---------------------------------------------------------------------------
// 1) fp32 -> bf16 conversion of activations
// ---------------------------------------------------------------------------
__global__ void moe_convert_kernel(const float* __restrict__ x,
                                   unsigned short* __restrict__ xb) {
    int i = (blockIdx.x * 256 + threadIdx.x) * 4;
    float4 v = *(const float4*)(x + i);
    unsigned int u0 = (unsigned int)f2bf(v.x) | ((unsigned int)f2bf(v.y) << 16);
    unsigned int u1 = (unsigned int)f2bf(v.z) | ((unsigned int)f2bf(v.w) << 16);
    *(uint2*)(xb + i) = make_uint2(u0, u1);
}

// ---------------------------------------------------------------------------
// 2) router: logits, softmax, top-2, normalized weights.  One wave per token.
// ---------------------------------------------------------------------------
__global__ void moe_router_kernel(const float* __restrict__ x,
                                  const float* __restrict__ rw,
                                  float* __restrict__ logits_out,
                                  int*   __restrict__ tokExp,
                                  float* __restrict__ tokW) {
    const int tid  = threadIdx.x;
    const int lane = tid & 31;
    const int t    = blockIdx.x * 8 + (tid >> 5);

    float acc[kE];
#pragma unroll
    for (int e = 0; e < kE; ++e) acc[e] = 0.f;

    const float* xp = x + (size_t)t * kH;
    for (int i = lane; i < kH; i += 32) {
        float xv = xp[i];
#pragma unroll
        for (int e = 0; e < kE; ++e) acc[e] += xv * rw[e * kH + i];
    }
#pragma unroll
    for (int m = 16; m; m >>= 1)
#pragma unroll
        for (int e = 0; e < kE; ++e) acc[e] += __shfl_xor(acc[e], m, 32);

    if (lane == 0) {
        float mx = acc[0];
#pragma unroll
        for (int e = 1; e < kE; ++e) mx = fmaxf(mx, acc[e]);
        float p[kE];
#pragma unroll
        for (int e = 0; e < kE; ++e) p[e] = expf(acc[e] - mx);
        int i0 = 0;
#pragma unroll
        for (int e = 1; e < kE; ++e) if (p[e] > p[i0]) i0 = e;       // lowest idx on tie
        int i1 = (i0 == 0) ? 1 : 0;
#pragma unroll
        for (int e = 0; e < kE; ++e) if (e != i0 && p[e] > p[i1]) i1 = e;
        float inv = 1.f / (p[i0] + p[i1]);
        tokExp[2 * t]     = i0;
        tokExp[2 * t + 1] = i1;
        tokW[2 * t]       = p[i0] * inv;
        tokW[2 * t + 1]   = p[i1] * inv;
#pragma unroll
        for (int e = 0; e < kE; ++e) logits_out[(size_t)t * kE + e] = acc[e];
    }
}

// ---------------------------------------------------------------------------
// 3) scatter: group (token, k) pairs by expert; segments padded to 64 rows.
// ---------------------------------------------------------------------------
__global__ void moe_scatter_kernel(const int* __restrict__ tokExp,
                                   const float* __restrict__ tokW,
                                   int*   __restrict__ pair_token,
                                   float* __restrict__ pair_w,
                                   int*   __restrict__ tokPos,
                                   int*   __restrict__ expOff) {
    __shared__ int cnt[kE];
    __shared__ int off[kE + 1];
    __shared__ int cur[kE];
    const int tid = threadIdx.x;
    if (tid < kE) cnt[tid] = 0;
    __syncthreads();
    for (int i = tid; i < kT * 2; i += 256) atomicAdd(&cnt[tokExp[i]], 1);
    __syncthreads();
    if (tid == 0) {
        int o = 0;
        for (int e = 0; e < kE; ++e) { off[e] = o; o += (cnt[e] + 63) & ~63; }
        off[kE] = o;
    }
    __syncthreads();
    if (tid < kE)     cur[tid]    = off[tid];
    if (tid < kE + 1) expOff[tid] = off[tid];
    for (int i = tid; i < PCAP; i += 256) { pair_token[i] = -1; pair_w[i] = 0.f; }
    __syncthreads();
    for (int i = tid; i < kT * 2; i += 256) {
        int e   = tokExp[i];
        int pos = atomicAdd(&cur[e], 1);
        pair_token[pos] = i >> 1;
        pair_w[pos]     = tokW[i];
        tokPos[i]       = pos;
    }
}

// ---------------------------------------------------------------------------
// 4) stage-1 GEMM:  G[r,f] = silu(X W1^T) * (X W3^T), bf16 output.
//    routed=1: rows gathered via pair_token, expert weights via expOff.
//    Pipelined: async A tile -> LDS[next], B tile global->reg while computing
//    from LDS[cur]; single barrier per k-step.  Pad rows clamp to token 0:
//    their outputs are finite garbage that the combine kernel never reads.
// ---------------------------------------------------------------------------
__global__ __launch_bounds__(256)
void moe_stage1_kernel(const unsigned short* __restrict__ xb,
                       const float* __restrict__ w1b,
                       const float* __restrict__ w3b,
                       unsigned short* __restrict__ gout,
                       const int* __restrict__ pair_token,
                       const int* __restrict__ expOff,
                       int routed) {
    __shared__ __align__(16) unsigned short As [2][BM * BK];
    __shared__ __align__(16) unsigned short B1s[2][BN * BK];
    __shared__ __align__(16) unsigned short B3s[2][BN * BK];

    const int tid = threadIdx.x;
    const int m0  = blockIdx.y * BM;
    const int n0  = blockIdx.x * BN;

    const float* w1 = w1b;
    const float* w3 = w3b;
    if (routed) {
        int e = 0;
#pragma unroll
        for (int i = 1; i < kE; ++i) if (m0 >= expOff[i]) e = i;
        size_t wofs = (size_t)e * kF * kH;
        w1 += wofs; w3 += wofs;
    }

    // per-thread A gather (fixed over k-loop); pad rows clamp to token 0
    const int arow  = m0 + (tid >> 2);
    const int apart = (tid & 3) * 8;                 // ushort offset in row
    int atok = routed ? pair_token[arow] : arow;
    if (atok < 0) atok = 0;
    const unsigned short* aptr = xb + (size_t)atok * kH + apart;
    const unsigned aslot[2] = {
        (unsigned)(uintptr_t)&As[0][(tid >> 2) * BK + apart],
        (unsigned)(uintptr_t)&As[1][(tid >> 2) * BK + apart] };

    const int brow = tid >> 1;                       // 0..127
    const int bcol = (tid & 1) * 16;                 // float offset in row
    const float* p1 = w1 + (size_t)(n0 + brow) * kH + bcol;
    const float* p3 = w3 + (size_t)(n0 + brow) * kH + bcol;

    const int wave = tid >> 5;
    const int wm = wave >> 2, wn = wave & 3;
    const int lane = tid & 31;
    const int lrow = lane & 15, lhi = lane >> 4;

    v8f z;
#pragma unroll
    for (int i = 0; i < 8; ++i) z[i] = 0.f;
    v8f acc1[2][2], acc3[2][2];
#pragma unroll
    for (int mi = 0; mi < 2; ++mi)
#pragma unroll
        for (int ni = 0; ni < 2; ++ni) { acc1[mi][ni] = z; acc3[mi][ni] = z; }

    // ---- prologue: fill buffer 0 ----
    async_ld_b128(aslot[0], aptr);
    {
        float4 f0 = ((const float4*)p1)[0], f1 = ((const float4*)p1)[1];
        float4 f2 = ((const float4*)p1)[2], f3 = ((const float4*)p1)[3];
        store_bf16x16(&B1s[0][brow * BK + bcol], f0, f1, f2, f3);
        f0 = ((const float4*)p3)[0]; f1 = ((const float4*)p3)[1];
        f2 = ((const float4*)p3)[2]; f3 = ((const float4*)p3)[3];
        store_bf16x16(&B3s[0][brow * BK + bcol], f0, f1, f2, f3);
    }
    wait_asynccnt0();
    __syncthreads();

    // ---- pipelined main loop: one barrier per k-step ----
    for (int it = 0; it < KSTEPS; ++it) {
        const int cur = it & 1, nxt = cur ^ 1;
        const bool more = (it + 1 < KSTEPS);
        const int k1 = (it + 1) * BK;

        float4 n1[4], n3[4];
        if (more) {
            async_ld_b128(aslot[nxt], aptr + k1);    // A tile k+1 -> LDS[nxt]
            const float* q1 = p1 + k1;
            const float* q3 = p3 + k1;
            __builtin_prefetch(q1 + BK, 0, 1);
            __builtin_prefetch(q3 + BK, 0, 1);
            n1[0] = ((const float4*)q1)[0]; n1[1] = ((const float4*)q1)[1];
            n1[2] = ((const float4*)q1)[2]; n1[3] = ((const float4*)q1)[3];
            n3[0] = ((const float4*)q3)[0]; n3[1] = ((const float4*)q3)[1];
            n3[2] = ((const float4*)q3)[2]; n3[3] = ((const float4*)q3)[3];
        }

        // fragments + WMMA from buffer[cur] (overlaps the loads above)
        BVec a[2], b1[2], b3[2];
#pragma unroll
        for (int mi = 0; mi < 2; ++mi) {
            const unsigned short* s = &As[cur][(wm * 32 + mi * 16 + lrow) * BK];
            a[mi].q[0] = *(const uint4*)(s + lhi * 8);
            a[mi].q[1] = *(const uint4*)(s + 16 + lhi * 8);
        }
#pragma unroll
        for (int ni = 0; ni < 2; ++ni) {
            const unsigned short* s1 = &B1s[cur][(wn * 32 + ni * 16 + lrow) * BK] + lhi * 8;
            b1[ni].q[0] = *(const uint4*)(s1);
            b1[ni].q[1] = *(const uint4*)(s1 + 16);
            const unsigned short* s3 = &B3s[cur][(wn * 32 + ni * 16 + lrow) * BK] + lhi * 8;
            b3[ni].q[0] = *(const uint4*)(s3);
            b3[ni].q[1] = *(const uint4*)(s3 + 16);
        }
#pragma unroll
        for (int mi = 0; mi < 2; ++mi)
#pragma unroll
            for (int ni = 0; ni < 2; ++ni) {
                acc1[mi][ni] = __builtin_amdgcn_wmma_f32_16x16x32_bf16(
                    false, a[mi].v, false, b1[ni].v, (short)0, acc1[mi][ni], false, false);
                acc3[mi][ni] = __builtin_amdgcn_wmma_f32_16x16x32_bf16(
                    false, a[mi].v, false, b3[ni].v, (short)0, acc3[mi][ni], false, false);
            }

        if (more) {
            store_bf16x16(&B1s[nxt][brow * BK + bcol], n1[0], n1[1], n1[2], n1[3]);
            store_bf16x16(&B3s[nxt][brow * BK + bcol], n3[0], n3[1], n3[2], n3[3]);
            wait_asynccnt0();                        // our async A write landed in LDS
        }
        __syncthreads();                             // publish [nxt]; retire reads of [cur]
    }

    // epilogue: silu(h1) * h3 -> bf16
#pragma unroll
    for (int mi = 0; mi < 2; ++mi)
#pragma unroll
        for (int ni = 0; ni < 2; ++ni)
#pragma unroll
            for (int r = 0; r < 8; ++r) {
                int row = m0 + wm * 32 + mi * 16 + lhi * 8 + r;
                int col = n0 + wn * 32 + ni * 16 + lrow;
                float v1 = acc1[mi][ni][r];
                float v3 = acc3[mi][ni][r];
                float g  = (v1 / (1.f + expf(-v1))) * v3;
                gout[(size_t)row * kF + col] = f2bf(g);
            }
}

// ---------------------------------------------------------------------------
// 5) stage-2 GEMM:  Y[r,h] = G W2^T   (f32 output), same pipeline
// ---------------------------------------------------------------------------
__global__ __launch_bounds__(256)
void moe_stage2_kernel(const unsigned short* __restrict__ gin,
                       const float* __restrict__ w2b,
                       float* __restrict__ out,
                       const int* __restrict__ expOff,
                       int routed) {
    __shared__ __align__(16) unsigned short As[2][BM * BK];
    __shared__ __align__(16) unsigned short Bs[2][BN * BK];

    const int tid = threadIdx.x;
    const int m0  = blockIdx.y * BM;
    const int n0  = blockIdx.x * BN;

    const float* w2 = w2b;
    if (routed) {
        int e = 0;
#pragma unroll
        for (int i = 1; i < kE; ++i) if (m0 >= expOff[i]) e = i;
        w2 += (size_t)e * kH * kF;
    }

    const int arow  = m0 + (tid >> 2);
    const int apart = (tid & 3) * 8;
    const unsigned short* aptr = gin + (size_t)arow * kF + apart;
    const unsigned aslot[2] = {
        (unsigned)(uintptr_t)&As[0][(tid >> 2) * BK + apart],
        (unsigned)(uintptr_t)&As[1][(tid >> 2) * BK + apart] };

    const int brow = tid >> 1;
    const int bcol = (tid & 1) * 16;
    const float* pw = w2 + (size_t)(n0 + brow) * kF + bcol;

    const int wave = tid >> 5;
    const int wm = wave >> 2, wn = wave & 3;
    const int lane = tid & 31;
    const int lrow = lane & 15, lhi = lane >> 4;

    v8f z;
#pragma unroll
    for (int i = 0; i < 8; ++i) z[i] = 0.f;
    v8f acc[2][2];
#pragma unroll
    for (int mi = 0; mi < 2; ++mi)
#pragma unroll
        for (int ni = 0; ni < 2; ++ni) acc[mi][ni] = z;

    // prologue
    async_ld_b128(aslot[0], aptr);
    {
        float4 f0 = ((const float4*)pw)[0], f1 = ((const float4*)pw)[1];
        float4 f2 = ((const float4*)pw)[2], f3 = ((const float4*)pw)[3];
        store_bf16x16(&Bs[0][brow * BK + bcol], f0, f1, f2, f3);
    }
    wait_asynccnt0();
    __syncthreads();

    for (int it = 0; it < KSTEPS; ++it) {
        const int cur = it & 1, nxt = cur ^ 1;
        const bool more = (it + 1 < KSTEPS);
        const int k1 = (it + 1) * BK;

        float4 nb[4];
        if (more) {
            async_ld_b128(aslot[nxt], aptr + k1);
            const float* q = pw + k1;
            __builtin_prefetch(q + BK, 0, 1);
            nb[0] = ((const float4*)q)[0]; nb[1] = ((const float4*)q)[1];
            nb[2] = ((const float4*)q)[2]; nb[3] = ((const float4*)q)[3];
        }

        BVec a[2], b[2];
#pragma unroll
        for (int mi = 0; mi < 2; ++mi) {
            const unsigned short* s = &As[cur][(wm * 32 + mi * 16 + lrow) * BK];
            a[mi].q[0] = *(const uint4*)(s + lhi * 8);
            a[mi].q[1] = *(const uint4*)(s + 16 + lhi * 8);
        }
#pragma unroll
        for (int ni = 0; ni < 2; ++ni) {
            const unsigned short* s = &Bs[cur][(wn * 32 + ni * 16 + lrow) * BK] + lhi * 8;
            b[ni].q[0] = *(const uint4*)(s);
            b[ni].q[1] = *(const uint4*)(s + 16);
        }
#pragma unroll
        for (int mi = 0; mi < 2; ++mi)
#pragma unroll
            for (int ni = 0; ni < 2; ++ni)
                acc[mi][ni] = __builtin_amdgcn_wmma_f32_16x16x32_bf16(
                    false, a[mi].v, false, b[ni].v, (short)0, acc[mi][ni], false, false);

        if (more) {
            store_bf16x16(&Bs[nxt][brow * BK + bcol], nb[0], nb[1], nb[2], nb[3]);
            wait_asynccnt0();
        }
        __syncthreads();
    }

#pragma unroll
    for (int mi = 0; mi < 2; ++mi)
#pragma unroll
        for (int ni = 0; ni < 2; ++ni)
#pragma unroll
            for (int r = 0; r < 8; ++r) {
                int row = m0 + wm * 32 + mi * 16 + lhi * 8 + r;
                int col = n0 + wn * 32 + ni * 16 + lrow;
                out[(size_t)row * kH + col] = acc[mi][ni][r];
            }
}

// ---------------------------------------------------------------------------
// 6) combine: out[t] = shared[t] + w0*Y[p0] + w1*Y[p1]
// ---------------------------------------------------------------------------
__global__ void moe_combine_kernel(float* __restrict__ out,
                                   const float* __restrict__ y,
                                   const int* __restrict__ tokPos,
                                   const float* __restrict__ tokW) {
    const int t = blockIdx.y;
    const int h = blockIdx.x * 256 + threadIdx.x;
    const int p0 = tokPos[2 * t], p1 = tokPos[2 * t + 1];
    const float w0 = tokW[2 * t], w1 = tokW[2 * t + 1];
    const size_t o = (size_t)t * kH + h;
    out[o] += w0 * y[(size_t)p0 * kH + h] + w1 * y[(size_t)p1 * kH + h];
}

// ---------------------------------------------------------------------------
// launch
// ---------------------------------------------------------------------------
extern "C" void kernel_launch(void* const* d_in, const int* in_sizes, int n_in,
                              void* d_out, int out_size, void* d_ws, size_t ws_size,
                              hipStream_t stream) {
    const float* x   = (const float*)d_in[0];
    const float* w1  = (const float*)d_in[1];
    const float* w2  = (const float*)d_in[2];
    const float* w3  = (const float*)d_in[3];
    const float* sw1 = (const float*)d_in[4];
    const float* sw2 = (const float*)d_in[5];
    const float* sw3 = (const float*)d_in[6];
    const float* rw  = (const float*)d_in[7];

    float* out    = (float*)d_out;
    float* logits = out + (size_t)kT * kH;

    constexpr size_t XB  = (size_t)kT * kH * 2;      // x in bf16
    constexpr size_t GS  = (size_t)kT * kF * 2;      // shared-expert G (bf16)
    constexpr size_t GR  = (size_t)PCAP * kF * 2;    // routed G (bf16)
    constexpr size_t YR  = (size_t)PCAP * kH * 4;    // routed Y (f32)
    constexpr size_t TE  = (size_t)kT * 2 * 4;       // tokExp / tokW / tokPos
    constexpr size_t PT  = (size_t)PCAP * 4;         // pair_token / pair_w
    constexpr size_t NEED = XB + GS + GR + YR + 3 * TE + 2 * PT + 64;
    if (ws_size < NEED) return;

    char* ws = (char*)d_ws;
    unsigned short* xb   = (unsigned short*)ws;        ws += XB;
    unsigned short* gsh  = (unsigned short*)ws;        ws += GS;
    unsigned short* grt  = (unsigned short*)ws;        ws += GR;
    float*          yrt  = (float*)ws;                 ws += YR;
    int*            tokExp = (int*)ws;                 ws += TE;
    float*          tokW   = (float*)ws;               ws += TE;
    int*            tokPos = (int*)ws;                 ws += TE;
    int*            pairT  = (int*)ws;                 ws += PT;
    float*          pairW  = (float*)ws;               ws += PT;
    int*            expOff = (int*)ws;

    moe_convert_kernel<<<(kT * kH / 4) / 256, 256, 0, stream>>>(x, xb);
    moe_router_kernel<<<kT / 8, 256, 0, stream>>>(x, rw, logits, tokExp, tokW);
    moe_scatter_kernel<<<1, 256, 0, stream>>>(tokExp, tokW, pairT, pairW, tokPos, expOff);
    moe_stage1_kernel<<<dim3(kF / BN, kT / BM), 256, 0, stream>>>(
        xb, sw1, sw3, gsh, nullptr, nullptr, 0);
    moe_stage2_kernel<<<dim3(kH / BN, kT / BM), 256, 0, stream>>>(
        gsh, sw2, out, nullptr, 0);
    moe_stage1_kernel<<<dim3(kF / BN, PCAP / BM), 256, 0, stream>>>(
        xb, w1, w3, grt, pairT, expOff, 1);
    moe_stage2_kernel<<<dim3(kH / BN, PCAP / BM), 256, 0, stream>>>(
        grt, w2, yrt, expOff, 1);
    moe_combine_kernel<<<dim3(kH / 256, kT), 256, 0, stream>>>(out, yrt, tokPos, tokW);
}
